// CausalSelfAttention_1700807050117
// MI455X (gfx1250) — matrix operand order repaired and checked by reading, not compiled, not run on hardware
//
#include <hip/hip_runtime.h>
#include <math.h>

typedef __bf16 bf16;
typedef __attribute__((ext_vector_type(16))) __bf16 v16bf;
typedef __attribute__((ext_vector_type(8)))  float  v8f;

#define NEG_INF (-1e30f)

// round-to-nearest-even fp32 -> bf16
static __device__ __forceinline__ bf16 f2bf(float f) {
  unsigned u = __builtin_bit_cast(unsigned, f);
  unsigned r = u + 0x7FFFu + ((u >> 16) & 1u);
  unsigned short h = (unsigned short)(r >> 16);
  return __builtin_bit_cast(bf16, h);
}
static __device__ __forceinline__ bf16 bf_zero() {
  return __builtin_bit_cast(bf16, (unsigned short)0);
}

// ---- WMMA fragment loaders (CDNA5 ISA 7.12.2 layouts, wave32) ----
// A 16x32 bf16: lane m = lane&15, hi = lane>>4.
//   hi=0: elems 0-7 = K0..7,  elems 8-15 = K16..23
//   hi=1: elems 0-7 = K8..15, elems 8-15 = K24..31
static __device__ __forceinline__ v16bf load_afrag(const bf16* __restrict__ p, int hi) {
  v16bf f;
  const bf16* p0 = p + hi * 8;
#pragma unroll
  for (int e = 0; e < 8; ++e) { f[e] = p0[e]; f[8 + e] = p0[16 + e]; }
  return f;
}
// B 32x16 bf16, lane col n = lane&15: same K order as A (hi=0: K0..7,K16..23; hi=1: K8..15,K24..31).
static __device__ __forceinline__ v16bf load_bfrag(const bf16* __restrict__ p, int hi) {
  v16bf f;
  const bf16* p0 = p + hi * 8;
#pragma unroll
  for (int e = 0; e < 8; ++e) { f[e] = p0[e]; f[8 + e] = p0[16 + e]; }
  return f;
}
static __device__ __forceinline__ v8f wmma_bf16(v16bf a, v16bf b, v8f c) {
  return __builtin_amdgcn_wmma_f32_16x16x32_bf16(false, a, false, b, (short)0, c, false, false);
}

// ---- cross-lane row reductions within each 16-lane half (matches C layout) ----
static __device__ __forceinline__ float rowmax16(float v) {
  v = fmaxf(v, __shfl_xor(v, 1, 32));
  v = fmaxf(v, __shfl_xor(v, 2, 32));
  v = fmaxf(v, __shfl_xor(v, 4, 32));
  v = fmaxf(v, __shfl_xor(v, 8, 32));
  return v;
}
static __device__ __forceinline__ float rowsum16(float v) {
  v += __shfl_xor(v, 1, 32);
  v += __shfl_xor(v, 2, 32);
  v += __shfl_xor(v, 4, 32);
  v += __shfl_xor(v, 8, 32);
  return v;
}

// online softmax update for one 16x16 score tile held as C fragment
static __device__ __forceinline__ void online_update(const v8f& c, float scale, bool diag,
                                                     int l4, int hi,
                                                     float* rmax, float* rsum) {
#pragma unroll
  for (int i = 0; i < 8; ++i) {
    const int mrow = i + 8 * hi;
    float s = c[i] * scale;
    if (diag && l4 > mrow) s = NEG_INF;
    const float tmax = rowmax16(s);
    const float mnew = fmaxf(rmax[i], tmax);
    const float e = rowsum16(__expf(s - mnew));
    rsum[i] = rsum[i] * __expf(rmax[i] - mnew) + e;
    rmax[i] = mnew;
  }
}

// compute one 16x16 score tile Q(16xD) x K^T, double-buffered D loop
static __device__ __forceinline__ v8f score_tile1(const bf16* __restrict__ qrow,
                                                  const bf16* __restrict__ krow,
                                                  int Dd, int hi) {
  v8f c = {};
  v16bf a = load_afrag(qrow, hi);
  v16bf b = load_bfrag(krow, hi);
  for (int d0 = 0; d0 < Dd; d0 += 32) {
    v16bf an, bn;
    const int dn = d0 + 32;
    if (dn < Dd) { an = load_afrag(qrow + dn, hi); bn = load_bfrag(krow + dn, hi); }
    else         { an = a; bn = b; }
    c = wmma_bf16(a, b, c);
    a = an; b = bn;
  }
  return c;
}

// compute four 16x16 score tiles sharing the A fragment, double-buffered
static __device__ __forceinline__ void score_tiles4(const bf16* __restrict__ qrow,
                                                    const bf16* __restrict__ k0,
                                                    const bf16* __restrict__ k1,
                                                    const bf16* __restrict__ k2,
                                                    const bf16* __restrict__ k3,
                                                    int Dd, int hi,
                                                    v8f& c0, v8f& c1, v8f& c2, v8f& c3) {
  v16bf a = load_afrag(qrow, hi);
  v16bf b0 = load_bfrag(k0, hi);
  v16bf b1 = load_bfrag(k1, hi);
  v16bf b2 = load_bfrag(k2, hi);
  v16bf b3 = load_bfrag(k3, hi);
  for (int d0 = 0; d0 < Dd; d0 += 32) {
    v16bf an, b0n, b1n, b2n, b3n;
    const int dn = d0 + 32;
    if (dn < Dd) {
      an  = load_afrag(qrow + dn, hi);
      b0n = load_bfrag(k0 + dn, hi);
      b1n = load_bfrag(k1 + dn, hi);
      b2n = load_bfrag(k2 + dn, hi);
      b3n = load_bfrag(k3 + dn, hi);
    } else { an = a; b0n = b0; b1n = b1; b2n = b2; b3n = b3; }
    c0 = wmma_bf16(a, b0, c0);
    c1 = wmma_bf16(a, b1, c1);
    c2 = wmma_bf16(a, b2, c2);
    c3 = wmma_bf16(a, b3, c3);
    a = an; b0 = b0n; b1 = b1n; b2 = b2n; b3 = b3n;
  }
}

// ---- fp32 -> bf16 conversion ----
__global__ void cvt_f32_bf16(const float* __restrict__ in, bf16* __restrict__ out, long n) {
  long i = (long)blockIdx.x * blockDim.x + threadIdx.x;
  long stride = (long)gridDim.x * blockDim.x;
  for (; i < n; i += stride) out[i] = f2bf(in[i]);
}

// ---- blocked bf16 WMMA GEMM: Y[M,N] = A[M,K] * B^T (+bias) ----
// A: [M,K] row-major. B: [N,K] row-major (contract along K, contiguous).
// Wave computes a 32x64 slab (2x4 tiles, 8 WMMAs per K-step, double-buffered).
// Workgroup = 8 waves (2 M x 4 N) -> 64 x 256 block tile.
// YT!=0: bf16 output stored transposed as [N,M].
__global__ void __launch_bounds__(256, 1)
gemm_bf16_nt(const bf16* __restrict__ A, const bf16* __restrict__ Bm,
             const float* __restrict__ bias,
             float* __restrict__ Yf, bf16* __restrict__ Ybf, int YT,
             int M, int N, int K,
             long strideA, long strideB, long strideY) {
  const int wave = threadIdx.x >> 5;
  const int lane = threadIdx.x & 31;
  const int l4 = lane & 15;
  const int hi = lane >> 4;
  const int wm = wave >> 2;
  const int wn = wave & 3;
  const int m0 = blockIdx.x * 64 + wm * 32;
  const int n0 = blockIdx.y * 256 + wn * 64;
  const int bz = blockIdx.z;

  const bf16* Ab = A + (long)bz * strideA;
  const bf16* Bb = Bm + (long)bz * strideB;
  const bf16* ar0 = Ab + (long)(m0 + l4) * K;
  const bf16* ar1 = ar0 + (long)16 * K;
  const bf16* br0 = Bb + (long)(n0 + l4) * K;
  const bf16* br1 = br0 + (long)16 * K;
  const bf16* br2 = br0 + (long)32 * K;
  const bf16* br3 = br0 + (long)48 * K;

  v8f acc[2][4];
  const v8f vz = {};
#pragma unroll
  for (int mi = 0; mi < 2; ++mi)
#pragma unroll
    for (int t = 0; t < 4; ++t) acc[mi][t] = vz;

  v16bf a0 = load_afrag(ar0, hi);
  v16bf a1 = load_afrag(ar1, hi);
  v16bf b0 = load_bfrag(br0, hi);
  v16bf b1 = load_bfrag(br1, hi);
  v16bf b2 = load_bfrag(br2, hi);
  v16bf b3 = load_bfrag(br3, hi);

  for (int k0 = 0; k0 < K; k0 += 32) {
    v16bf a0n, a1n, b0n, b1n, b2n, b3n;
    const int kn = k0 + 32;
    if (kn < K) {
      a0n = load_afrag(ar0 + kn, hi);
      a1n = load_afrag(ar1 + kn, hi);
      b0n = load_bfrag(br0 + kn, hi);
      b1n = load_bfrag(br1 + kn, hi);
      b2n = load_bfrag(br2 + kn, hi);
      b3n = load_bfrag(br3 + kn, hi);
    } else { a0n = a0; a1n = a1; b0n = b0; b1n = b1; b2n = b2; b3n = b3; }
    acc[0][0] = wmma_bf16(a0, b0, acc[0][0]);
    acc[0][1] = wmma_bf16(a0, b1, acc[0][1]);
    acc[0][2] = wmma_bf16(a0, b2, acc[0][2]);
    acc[0][3] = wmma_bf16(a0, b3, acc[0][3]);
    acc[1][0] = wmma_bf16(a1, b0, acc[1][0]);
    acc[1][1] = wmma_bf16(a1, b1, acc[1][1]);
    acc[1][2] = wmma_bf16(a1, b2, acc[1][2]);
    acc[1][3] = wmma_bf16(a1, b3, acc[1][3]);
    a0 = a0n; a1 = a1n; b0 = b0n; b1 = b1n; b2 = b2n; b3 = b3n;
  }

  float bv[4];
#pragma unroll
  for (int t = 0; t < 4; ++t) bv[t] = bias ? bias[n0 + t * 16 + l4] : 0.0f;
  float* Yfb = Yf ? (Yf + (long)bz * strideY) : (float*)0;
  bf16* Ybb = Ybf ? (Ybf + (long)bz * strideY) : (bf16*)0;
#pragma unroll
  for (int mi = 0; mi < 2; ++mi) {
#pragma unroll
    for (int t = 0; t < 4; ++t) {
#pragma unroll
      for (int i = 0; i < 8; ++i) {
        const int m = m0 + mi * 16 + i + 8 * hi;   // C layout: row = vgpr + 8*(lane>=16)
        const int n = n0 + t * 16 + l4;            //           col = lane&15
        const float v = acc[mi][t][i] + bv[t];
        if (Yfb) Yfb[(long)m * N + n] = v;
        if (Ybb) {
          const long idx = YT ? ((long)n * M + m) : ((long)m * N + n);
          Ybb[idx] = f2bf(v);
        }
      }
    }
  }
}

// ---- pass 1: partial online-softmax stats; 8 key-stripes per (b,qblk) ----
// grid (S/16, 1, B), block 256: wave = key stripe 0..7
__global__ void __launch_bounds__(256, 1)
attn_stats4(const bf16* __restrict__ Q, const bf16* __restrict__ Kt,
            float* __restrict__ mpart, float* __restrict__ lpart,
            int S, int Dd, float scale) {
  const int split = threadIdx.x >> 5;
  const int lane = threadIdx.x & 31;
  const int l4 = lane & 15;
  const int hi = lane >> 4;
  const int qblk = blockIdx.x;
  const int b = blockIdx.z;
  const int KT8 = (S / 16) / 8;                 // 16 key tiles per stripe
  const int jbeg = split * KT8;
  int jend = qblk + 1;
  if (jend > jbeg + KT8) jend = jbeg + KT8;

  float rmax[8], rsum[8];
#pragma unroll
  for (int i = 0; i < 8; ++i) { rmax[i] = NEG_INF; rsum[i] = 0.0f; }

  if (jbeg < jend) {
    const bf16* Qb = Q + (long)b * S * Dd;
    const bf16* Kb = Kt + (long)b * S * Dd;
    const bf16* qrow = Qb + (long)(qblk * 16 + l4) * Dd;
    int j = jbeg;
    for (; j + 3 < jend; j += 4) {
      const bf16* kr = Kb + (long)(j * 16 + l4) * Dd;
      v8f c0 = {}, c1 = {}, c2 = {}, c3 = {};
      score_tiles4(qrow, kr, kr + (long)16 * Dd, kr + (long)32 * Dd, kr + (long)48 * Dd,
                   Dd, hi, c0, c1, c2, c3);
      online_update(c0, scale, (j + 0) == qblk, l4, hi, rmax, rsum);
      online_update(c1, scale, (j + 1) == qblk, l4, hi, rmax, rsum);
      online_update(c2, scale, (j + 2) == qblk, l4, hi, rmax, rsum);
      online_update(c3, scale, (j + 3) == qblk, l4, hi, rmax, rsum);
    }
    for (; j < jend; ++j) {
      const bf16* kr = Kb + (long)(j * 16 + l4) * Dd;
      v8f c = score_tile1(qrow, kr, Dd, hi);
      online_update(c, scale, j == qblk, l4, hi, rmax, rsum);
    }
  }
  if (l4 == 0) {
#pragma unroll
    for (int i = 0; i < 8; ++i) {
      const int r = qblk * 16 + i + 8 * hi;
      mpart[((long)b * S + r) * 8 + split] = rmax[i];
      lpart[((long)b * S + r) * 8 + split] = rsum[i];
    }
  }
}

// merge 8 partial (m,l) pairs per row (log-sum-exp merge)
__global__ void stats_combine(const float* __restrict__ mpart, const float* __restrict__ lpart,
                              float* __restrict__ mbuf, float* __restrict__ lbuf, long nrows) {
  long r = (long)blockIdx.x * blockDim.x + threadIdx.x;
  if (r >= nrows) return;
  float m = NEG_INF;
#pragma unroll
  for (int s = 0; s < 8; ++s) m = fmaxf(m, mpart[r * 8 + s]);
  float l = 0.0f;
#pragma unroll
  for (int s = 0; s < 8; ++s) l += lpart[r * 8 + s] * __expf(mpart[r * 8 + s] - m);
  mbuf[r] = m;
  lbuf[r] = l;
}

// ---- pass 2: normalized probabilities P (bf16, zeros above diagonal) ----
// grid (S/64/8, S/16, B): wave handles 4 consecutive key tiles of one q-block
__global__ void __launch_bounds__(256, 1)
attn_probs4(const bf16* __restrict__ Q, const bf16* __restrict__ Kt,
            const float* __restrict__ mbuf, const float* __restrict__ lbuf,
            bf16* __restrict__ P, int S, int Dd, float scale) {
  const int wave = threadIdx.x >> 5;
  const int lane = threadIdx.x & 31;
  const int l4 = lane & 15;
  const int hi = lane >> 4;
  const int kgrp = blockIdx.x * 8 + wave;       // 0..S/64-1
  const int qblk = blockIdx.y;
  const int b = blockIdx.z;
  const int kblk0 = kgrp * 4;

  bf16* Pt = P + ((long)b * S + (long)qblk * 16) * S + (long)kblk0 * 16;
  if (kblk0 > qblk) {                            // entirely above diagonal -> zeros
    const bf16 z = bf_zero();
#pragma unroll
    for (int t = 0; t < 4; ++t)
#pragma unroll
      for (int i = 0; i < 8; ++i)
        Pt[(long)(i + 8 * hi) * S + t * 16 + l4] = z;
    return;
  }

  const bf16* qrow = Q + ((long)b * S + qblk * 16 + l4) * Dd;
  const bf16* kr = Kt + ((long)b * S + kblk0 * 16 + l4) * Dd;
  v8f c0 = {}, c1 = {}, c2 = {}, c3 = {};
  score_tiles4(qrow, kr, kr + (long)16 * Dd, kr + (long)32 * Dd, kr + (long)48 * Dd,
               Dd, hi, c0, c1, c2, c3);

  float mm[8], ll[8];
#pragma unroll
  for (int i = 0; i < 8; ++i) {
    const int r = qblk * 16 + i + 8 * hi;
    mm[i] = mbuf[(long)b * S + r];
    ll[i] = lbuf[(long)b * S + r];
  }

  const v8f* cc[4] = { &c0, &c1, &c2, &c3 };
#pragma unroll
  for (int t = 0; t < 4; ++t) {
    const int kblk = kblk0 + t;
    if (kblk > qblk) {
      const bf16 z = bf_zero();
#pragma unroll
      for (int i = 0; i < 8; ++i) Pt[(long)(i + 8 * hi) * S + t * 16 + l4] = z;
    } else {
      const bool diag = (kblk == qblk);
#pragma unroll
      for (int i = 0; i < 8; ++i) {
        const int mrow = i + 8 * hi;
        float s = (*cc[t])[i] * scale;
        if (diag && l4 > mrow) s = NEG_INF;
        const float p = __expf(s - mm[i]) / ll[i];
        Pt[(long)mrow * S + t * 16 + l4] = f2bf(p);
      }
    }
  }
}

extern "C" void kernel_launch(void* const* d_in, const int* in_sizes, int n_in,
                              void* d_out, int out_size, void* d_ws, size_t ws_size,
                              hipStream_t stream) {
  (void)in_sizes; (void)n_in; (void)out_size; (void)ws_size;
  const int Bsz = 4, S = 2048, Dd = 1024;
  const long M = (long)Bsz * S;   // 8192
  const float scale = 1.0f / sqrtf((float)Dd);

  const float* x  = (const float*)d_in[0];
  const float* Wq = (const float*)d_in[1];
  const float* bq = (const float*)d_in[2];
  const float* Wk = (const float*)d_in[3];
  const float* bk = (const float*)d_in[4];
  const float* Wv = (const float*)d_in[5];
  const float* bv = (const float*)d_in[6];
  const float* Wp = (const float*)d_in[7];
  const float* bp = (const float*)d_in[8];

  char* w = (char*)d_ws;
  size_t off = 0;
  auto carve = [&](size_t bytes) { void* p = w + off; off += (bytes + 255) & ~(size_t)255; return p; };
  bf16* x_bf  = (bf16*)carve((size_t)M * Dd * 2);
  bf16* wq_bf = (bf16*)carve((size_t)Dd * Dd * 2);
  bf16* wk_bf = (bf16*)carve((size_t)Dd * Dd * 2);
  bf16* wv_bf = (bf16*)carve((size_t)Dd * Dd * 2);
  bf16* wp_bf = (bf16*)carve((size_t)Dd * Dd * 2);
  bf16* q_bf  = (bf16*)carve((size_t)M * Dd * 2);
  bf16* k_bf  = (bf16*)carve((size_t)M * Dd * 2);
  bf16* vt_bf = (bf16*)carve((size_t)M * Dd * 2);          // V^T per batch: [B][D][S]
  bf16* o_bf  = (bf16*)carve((size_t)M * Dd * 2);
  bf16* Pbuf  = (bf16*)carve((size_t)Bsz * S * S * 2);
  float* mbuf = (float*)carve((size_t)M * 4);
  float* lbuf = (float*)carve((size_t)M * 4);
  float* mpart = (float*)carve((size_t)M * 8 * 4);
  float* lpart = (float*)carve((size_t)M * 8 * 4);

  // 1) fp32 -> bf16
  cvt_f32_bf16<<<1024, 256, 0, stream>>>(x, x_bf, M * Dd);
  cvt_f32_bf16<<<512, 256, 0, stream>>>(Wq, wq_bf, (long)Dd * Dd);
  cvt_f32_bf16<<<512, 256, 0, stream>>>(Wk, wk_bf, (long)Dd * Dd);
  cvt_f32_bf16<<<512, 256, 0, stream>>>(Wv, wv_bf, (long)Dd * Dd);
  cvt_f32_bf16<<<512, 256, 0, stream>>>(Wp, wp_bf, (long)Dd * Dd);

  // 2) projections (64x256 workgroup tiles)
  dim3 gproj((unsigned)(M / 64), (unsigned)(Dd / 256), 1);
  gemm_bf16_nt<<<gproj, 256, 0, stream>>>(x_bf, wq_bf, bq, (float*)0, q_bf, 0,
                                          (int)M, Dd, Dd, 0, 0, 0);
  gemm_bf16_nt<<<gproj, 256, 0, stream>>>(x_bf, wk_bf, bk, (float*)0, k_bf, 0,
                                          (int)M, Dd, Dd, 0, 0, 0);
  // V projection batched, bf16 result stored transposed -> V^T[b][n][s]
  dim3 gv((unsigned)(S / 64), (unsigned)(Dd / 256), (unsigned)Bsz);
  gemm_bf16_nt<<<gv, 256, 0, stream>>>(x_bf, wv_bf, bv, (float*)0, vt_bf, 1,
                                       S, Dd, Dd, (long)S * Dd, 0, (long)S * Dd);

  // 3) softmax stats: 8 key-stripes per q-block, then combine
  attn_stats4<<<dim3(S / 16, 1, Bsz), 256, 0, stream>>>(q_bf, k_bf, mpart, lpart, S, Dd, scale);
  stats_combine<<<(unsigned)((M + 255) / 256), 256, 0, stream>>>(mpart, lpart, mbuf, lbuf, M);

  // 4) probabilities
  attn_probs4<<<dim3(S / 64 / 8, S / 16, Bsz), 256, 0, stream>>>(q_bf, k_bf, mbuf, lbuf, Pbuf,
                                                                 S, Dd, scale);

  // 5) O = P @ V  (B operand = V^T rows, K-contiguous)
  dim3 gpv((unsigned)(S / 64), (unsigned)(Dd / 256), (unsigned)Bsz);
  gemm_bf16_nt<<<gpv, 256, 0, stream>>>(Pbuf, vt_bf, (const float*)0, (float*)0, o_bf, 0,
                                        S, Dd, S,
                                        (long)S * S, (long)Dd * S, (long)S * Dd);

  // 6) output projection -> fp32 d_out
  dim3 gout((unsigned)(M / 64), (unsigned)(Dd / 256), 1);
  gemm_bf16_nt<<<gout, 256, 0, stream>>>(o_bf, wp_bf, bp, (float*)d_out, (bf16*)0, 0,
                                         (int)M, Dd, Dd, 0, 0, 0);
}
